// MambaBlock_23338852286957
// MI455X (gfx1250) — matrix operand order repaired and checked by reading, not compiled, not run on hardware
//
#include <hip/hip_runtime.h>

// ---------------------------------------------------------------------------
// Mamba block (2 layers) for MI455X / gfx1250, wave32, WMMA bf16.
// B=2, T=1024, H=256, D=1024, N=16, R=16, K=4
// ---------------------------------------------------------------------------

#define EPSV 1e-5f

constexpr int CB = 2, CT = 1024, CH = 256, CD = 1024, CN = 16, CR = 16, CK = 4;
constexpr int BT = CB * CT;            // 2048 rows (M dimension everywhere)
constexpr int CE = CR + 2 * CN;        // 48 (xproj output width)

typedef __attribute__((ext_vector_type(16))) __bf16 v16bf;
typedef __attribute__((ext_vector_type(8)))  __bf16 v8bf;
typedef __attribute__((ext_vector_type(8)))  float  v8f;

#define AF_SHUF(lo, hi) __builtin_shufflevector((lo), (hi), \
    0, 1, 2, 3, 4, 5, 6, 7, 8, 9, 10, 11, 12, 13, 14, 15)

// --------------------------- block reduction -------------------------------
__device__ __forceinline__ float blk_sum256(float v) {
    __shared__ float red[256];
    int t = threadIdx.x;
    red[t] = v;
    __syncthreads();
    for (int s = 128; s > 0; s >>= 1) {
        if (t < s) red[t] += red[t + s];
        __syncthreads();
    }
    float r = red[0];
    __syncthreads();
    return r;
}

// ---------------- weight cast+transpose: W (N,K) f32 -> WT (K,N) bf16 -------
__global__ void k_cast_transpose(const float* __restrict__ W,
                                 __bf16* __restrict__ WT, int N, int K) {
    int idx = blockIdx.x * blockDim.x + threadIdx.x;
    if (idx >= N * K) return;
    int n = idx / K;
    int k = idx - n * K;
    WT[(size_t)k * N + n] = (__bf16)W[idx];
}

// ------------------------------ layer norm ----------------------------------
__global__ void k_layernorm(const float* __restrict__ x,
                            const float* __restrict__ g,
                            const float* __restrict__ b,
                            float* __restrict__ out) {
    int row = blockIdx.x, t = threadIdx.x;
    float v = x[(size_t)row * CH + t];
    float m = blk_sum256(v) * (1.0f / CH);
    float d = v - m;
    float var = blk_sum256(d * d) * (1.0f / CH);
    out[(size_t)row * CH + t] = d * rsqrtf(var + EPSV) * g[t] + b[t];
}

// --------------------- rms norm + cast to bf16 ------------------------------
__global__ void k_rms_cast(const float* __restrict__ h,
                           const float* __restrict__ w,
                           __bf16* __restrict__ hn16) {
    int row = blockIdx.x, t = threadIdx.x;
    float v = h[(size_t)row * CH + t];
    float ms = blk_sum256(v * v) * (1.0f / CH);
    hn16[(size_t)row * CH + t] = (__bf16)(v * rsqrtf(ms + EPSV) * w[t]);
}

// ---------------------------------------------------------------------------
// WMMA GEMM: C[M,N] = A[M,K] (bf16, row-major) * WT[K,N] (bf16)  (+ residual)
// 128 threads = 4 waves; each wave computes one 16 x (NT*16) strip of C.
//
// Ping-pong software pipeline, unrolled by 2 so the two fragment register
// sets alternate roles (no rotation copies). Requires K/32 even, which holds
// for all call sites (K=256 -> 8 steps, K=1024 -> 32 steps):
//   prologue: load step0 -> set0, step1 -> set1
//   loop:     wmma(set0); refill set0 <- step s+2;
//             wmma(set1); refill set1 <- step s+3;
//   tail:     wmma(set0); wmma(set1)
// Each WMMA group waits (partial s_wait_loadcnt) on loads issued one
// half-iteration earlier, overlapping VMEM with the XDL pipe.
//
// A fragment (16-bit A 16x32): lane group 0 -> K{0..7,16..23}, group 1 ->
// K{8..15,24..31}: two adjacent b128 loads per lane fold into v[a:a+7].
// B fragment: lane&15 -> column n, same K order as A (group 0 -> K{0..7,16..23},
// group 1 -> K{8..15,24..31}): strided gather of WT per lane.
// C layout: VGPR r -> row r (lanes 0-15) / r+8 (lanes 16-31), lane&15 -> col.
// ---------------------------------------------------------------------------
template <int NT, bool ADD_RES>
__global__ __launch_bounds__(128)
void k_gemm_wmma(const __bf16* __restrict__ A,
                 const __bf16* __restrict__ WT,
                 float* C, const float* Cres,
                 int M, int N, int K) {
    const int lane = threadIdx.x & 31;
    const int wave = threadIdx.x >> 5;
    const int rowBase = (blockIdx.y * 4 + wave) * 16;
    const int colBase = blockIdx.x * (NT * 16);

    const int g  = lane >> 4;       // lane group (K split)
    const int mr = lane & 15;       // row within tile (A) / col (B,C)
    const int row = rowBase + mr;

    v8f acc[NT];
    #pragma unroll
    for (int t = 0; t < NT; ++t) acc[t] = v8f{};

    const __bf16* pa = A + (size_t)row * K + g * 8;                // += 32
    const __bf16* pb = WT + (size_t)(g * 8) * N + colBase + mr;    // += 32*N
    const size_t pbStep = (size_t)32 * N;
    const int steps = K >> 5;                 // even (8 or 32) at all call sites

    // ---- prologue: step0 -> set0, step1 -> set1 ----
    v8bf alo0 = *(const v8bf*)pa;
    v8bf ahi0 = *(const v8bf*)(pa + 16);
    v16bf b0[NT];
    #pragma unroll
    for (int t = 0; t < NT; ++t) for (int s = 0; s < 16; ++s) b0[t][s] = pb[(size_t)(s + (s & 8)) * N + t * 16];

    pa += 32; pb += pbStep;
    v8bf alo1 = *(const v8bf*)pa;
    v8bf ahi1 = *(const v8bf*)(pa + 16);
    v16bf b1[NT];
    #pragma unroll
    for (int t = 0; t < NT; ++t) for (int s = 0; s < 16; ++s) b1[t][s] = pb[(size_t)(s + (s & 8)) * N + t * 16];

    // ---- steady state: 2 steps per iteration, ping-pong sets ----
    for (int s = 0; s < steps - 2; s += 2) {
        {   // compute step s (set0)
            v16bf af = AF_SHUF(alo0, ahi0);
            #pragma unroll
            for (int t = 0; t < NT; ++t)
                acc[t] = __builtin_amdgcn_wmma_f32_16x16x32_bf16(
                    false, af, false, b0[t], (short)0, acc[t], false, false);
        }
        pa += 32; pb += pbStep;               // -> step s+2
        __builtin_prefetch(pa + 64, 0, 3);    // speculative; OOB dropped
        alo0 = *(const v8bf*)pa;
        ahi0 = *(const v8bf*)(pa + 16);
        #pragma unroll
        for (int t = 0; t < NT; ++t) for (int s = 0; s < 16; ++s) b0[t][s] = pb[(size_t)(s + (s & 8)) * N + t * 16];

        {   // compute step s+1 (set1)
            v16bf af = AF_SHUF(alo1, ahi1);
            #pragma unroll
            for (int t = 0; t < NT; ++t)
                acc[t] = __builtin_amdgcn_wmma_f32_16x16x32_bf16(
                    false, af, false, b1[t], (short)0, acc[t], false, false);
        }
        pa += 32; pb += pbStep;               // -> step s+3
        alo1 = *(const v8bf*)pa;
        ahi1 = *(const v8bf*)(pa + 16);
        #pragma unroll
        for (int t = 0; t < NT; ++t) for (int s = 0; s < 16; ++s) b1[t][s] = pb[(size_t)(s + (s & 8)) * N + t * 16];
    }

    // ---- tail: steps-2 (set0), steps-1 (set1) ----
    {
        v16bf af = AF_SHUF(alo0, ahi0);
        #pragma unroll
        for (int t = 0; t < NT; ++t)
            acc[t] = __builtin_amdgcn_wmma_f32_16x16x32_bf16(
                false, af, false, b0[t], (short)0, acc[t], false, false);
    }
    {
        v16bf af = AF_SHUF(alo1, ahi1);
        #pragma unroll
        for (int t = 0; t < NT; ++t)
            acc[t] = __builtin_amdgcn_wmma_f32_16x16x32_bf16(
                false, af, false, b1[t], (short)0, acc[t], false, false);
    }

    #pragma unroll
    for (int t = 0; t < NT; ++t) {
        int col = colBase + t * 16 + mr;
        #pragma unroll
        for (int r = 0; r < 8; ++r) {
            int orow = rowBase + r + g * 8;
            size_t idx = (size_t)orow * N + col;
            float v = acc[t][r];
            if (ADD_RES) v += Cres[idx];
            C[idx] = v;
        }
    }
}

// --------------- depthwise causal conv (K=4) + bias + SiLU ------------------
__global__ void k_conv_silu(const float* __restrict__ proj,   // (BT, 2D)
                            const float* __restrict__ cw,     // (D, K)
                            const float* __restrict__ cb,     // (D)
                            float* __restrict__ u,            // (BT, D) f32
                            __bf16* __restrict__ u16) {       // (BT, D) bf16
    int tid = blockIdx.x * blockDim.x + threadIdx.x;
    if (tid >= BT * CD) return;
    int d = tid & (CD - 1);
    int bt = tid >> 10;               // CD = 1024
    int t = bt & (CT - 1);
    int b = bt >> 10;                 // CT = 1024
    float s = cb[d];
    #pragma unroll
    for (int k = 0; k < CK; ++k) {
        int tt = t - (CK - 1) + k;
        if (tt >= 0) {
            size_t pidx = ((size_t)(b * CT + tt)) * (2 * CD) + d;
            s += cw[d * CK + k] * proj[pidx];
        }
    }
    float sv = s / (1.0f + __expf(-s));       // silu
    u[tid]   = sv;
    u16[tid] = (__bf16)sv;
}

// --------- dt = softplus(ssm[:, :R] @ dtproj_w^T + dtproj_b)  (K=16) --------
__global__ void k_dtproj(const float* __restrict__ ssm,   // (BT, 48)
                         const float* __restrict__ dtw,   // (D, R)
                         const float* __restrict__ dtb,   // (D)
                         float* __restrict__ dt) {        // (BT, D)
    int tid = blockIdx.x * blockDim.x + threadIdx.x;
    if (tid >= BT * CD) return;
    int d = tid & (CD - 1);
    int bt = tid >> 10;
    const float* sr = ssm + (size_t)bt * CE;
    const float* wr = dtw + (size_t)d * CR;
    float z = dtb[d];
    #pragma unroll
    for (int r = 0; r < CR; ++r) z += sr[r] * wr[r];
    dt[tid] = (z > 20.0f) ? z : __logf(1.0f + __expf(z));
}

// ------------------------- selective scan (serial in T) ---------------------
// One thread per (b,d); 16 states + 16 A values live in registers.
__global__ void k_scan(const float* __restrict__ u,     // (BT, D)
                       const float* __restrict__ dt,    // (BT, D)
                       const float* __restrict__ ssm,   // (BT, 48): [R | B | C]
                       const float* __restrict__ A_log, // (D, N)
                       const float* __restrict__ Dp,    // (D)
                       float* __restrict__ y) {         // (BT, D)
    int gid = blockIdx.x * blockDim.x + threadIdx.x;
    if (gid >= CB * CD) return;
    int d = gid & (CD - 1);
    int b = gid >> 10;

    float Ad[CN];
    #pragma unroll
    for (int n = 0; n < CN; ++n) Ad[n] = -__expf(A_log[(size_t)d * CN + n]);
    float s[CN];
    #pragma unroll
    for (int n = 0; n < CN; ++n) s[n] = 0.0f;
    const float Dpv = Dp[d];

    for (int t = 0; t < CT; ++t) {
        size_t bt = (size_t)b * CT + t;
        float dtv = dt[bt * CD + d];
        float uv  = u[bt * CD + d];
        const float* Bp = ssm + bt * CE + CR;       // B (wave-uniform)
        const float* Cp = Bp + CN;                  // C
        float du = dtv * uv;
        float yacc = 0.0f;
        #pragma unroll
        for (int n = 0; n < CN; ++n) {
            float dA = __expf(dtv * Ad[n]);
            s[n] = dA * s[n] + du * Bp[n];
            yacc += s[n] * Cp[n];
        }
        y[bt * CD + d] = yacc + uv * Dpv;
    }
}

// ------------------------- y * silu(gate) -> bf16 ---------------------------
__global__ void k_gate(const float* __restrict__ y,     // (BT, D)
                       const float* __restrict__ proj,  // (BT, 2D), gate at +D
                       __bf16* __restrict__ yg16) {
    int tid = blockIdx.x * blockDim.x + threadIdx.x;
    if (tid >= BT * CD) return;
    int d = tid & (CD - 1);
    int bt = tid >> 10;
    float gg = proj[(size_t)bt * (2 * CD) + CD + d];
    float sg = gg / (1.0f + __expf(-gg));
    yg16[tid] = (__bf16)(y[tid] * sg);
}

// -------------- final: rms_norm(h, normf_w) then layer_norm -----------------
__global__ void k_final(const float* __restrict__ h,
                        const float* __restrict__ nfw,
                        const float* __restrict__ pg,
                        const float* __restrict__ pb,
                        float* __restrict__ out) {
    int row = blockIdx.x, t = threadIdx.x;
    float v = h[(size_t)row * CH + t];
    float ms = blk_sum256(v * v) * (1.0f / CH);
    float r = v * rsqrtf(ms + EPSV) * nfw[t];
    float m = blk_sum256(r) * (1.0f / CH);
    float d = r - m;
    float var = blk_sum256(d * d) * (1.0f / CH);
    out[(size_t)row * CH + t] = d * rsqrtf(var + EPSV) * pg[t] + pb[t];
}

// ---------------------------------------------------------------------------
extern "C" void kernel_launch(void* const* d_in, const int* in_sizes, int n_in,
                              void* d_out, int out_size, void* d_ws, size_t ws_size,
                              hipStream_t stream) {
    const float* x        = (const float*)d_in[0];
    const float* pre_g    = (const float*)d_in[1];
    const float* pre_b    = (const float*)d_in[2];
    const float* post_g   = (const float*)d_in[3];
    const float* post_b   = (const float*)d_in[4];
    const float* norm_w   = (const float*)d_in[5];   // (2, H)
    const float* inproj_w = (const float*)d_in[6];   // (2, 2D, H)
    const float* conv_w   = (const float*)d_in[7];   // (2, D, 1, K)
    const float* conv_b   = (const float*)d_in[8];   // (2, D)
    const float* xproj_w  = (const float*)d_in[9];   // (2, 48, D)
    const float* dtproj_w = (const float*)d_in[10];  // (2, D, R)
    const float* dtproj_b = (const float*)d_in[11];  // (2, D)
    const float* A_log    = (const float*)d_in[12];  // (2, D, N)
    const float* D_param  = (const float*)d_in[13];  // (2, D)
    const float* outproj_w= (const float*)d_in[14];  // (2, H, D)
    const float* normf_w  = (const float*)d_in[15];  // (H)

    // ------------------- workspace layout (256B aligned) --------------------
    char* ws = (char*)d_ws;
    auto align256 = [](size_t v) { return (v + 255) & ~(size_t)255; };
    size_t off = 0;
    auto take = [&](size_t bytes) { size_t o = off; off = align256(off + bytes); return o; };

    float*  h     = (float*) (ws + take((size_t)BT * CH * 4));
    __bf16* hn16  = (__bf16*)(ws + take((size_t)BT * CH * 2));
    float*  proj  = (float*) (ws + take((size_t)BT * 2 * CD * 4));
    float*  u     = (float*) (ws + take((size_t)BT * CD * 4));
    __bf16* u16   = (__bf16*)(ws + take((size_t)BT * CD * 2));
    float*  ssm   = (float*) (ws + take((size_t)BT * CE * 4));
    float*  dtv   = (float*) (ws + take((size_t)BT * CD * 4));
    float*  yv    = (float*) (ws + take((size_t)BT * CD * 4));
    __bf16* yg16  = (__bf16*)(ws + take((size_t)BT * CD * 2));
    __bf16* wtin  = (__bf16*)(ws + take((size_t)2 * CH * 2 * CD * 2));
    __bf16* wtx   = (__bf16*)(ws + take((size_t)2 * CD * CE * 2));
    __bf16* wtout = (__bf16*)(ws + take((size_t)2 * CD * CH * 2));
    (void)ws_size; (void)in_sizes; (void)n_in; (void)out_size;

    // ----------------- weight casts + transposes (per call) -----------------
    for (int i = 0; i < 2; ++i) {
        { int n = 2 * CD * CH;
          k_cast_transpose<<<(n + 255) / 256, 256, 0, stream>>>(
              inproj_w + (size_t)i * n, wtin + (size_t)i * n, 2 * CD, CH); }
        { int n = CE * CD;
          k_cast_transpose<<<(n + 255) / 256, 256, 0, stream>>>(
              xproj_w + (size_t)i * n, wtx + (size_t)i * n, CE, CD); }
        { int n = CH * CD;
          k_cast_transpose<<<(n + 255) / 256, 256, 0, stream>>>(
              outproj_w + (size_t)i * n, wtout + (size_t)i * n, CH, CD); }
    }

    // ----------------------------- forward ----------------------------------
    k_layernorm<<<BT, 256, 0, stream>>>(x, pre_g, pre_b, h);

    const int nElem = BT * CD;
    const int nBlk  = (nElem + 255) / 256;

    for (int i = 0; i < 2; ++i) {
        k_rms_cast<<<BT, 256, 0, stream>>>(h, norm_w + (size_t)i * CH, hn16);

        // inproj: (BT,2D) = hn16(BT,H) x wtin(H,2D)   N=2048 -> NT=4
        { dim3 g((2 * CD) / 64, BT / 64);
          k_gemm_wmma<4, false><<<g, 128, 0, stream>>>(
              hn16, wtin + (size_t)i * 2 * CD * CH, proj, nullptr,
              BT, 2 * CD, CH); }

        k_conv_silu<<<nBlk, 256, 0, stream>>>(
            proj, conv_w + (size_t)i * CD * CK, conv_b + (size_t)i * CD, u, u16);

        // xproj: (BT,48) = u16(BT,D) x wtx(D,48)      N=48 -> NT=3, 1 column
        { dim3 g(1, BT / 64);
          k_gemm_wmma<3, false><<<g, 128, 0, stream>>>(
              u16, wtx + (size_t)i * CD * CE, ssm, nullptr,
              BT, CE, CD); }

        k_dtproj<<<nBlk, 256, 0, stream>>>(
            ssm, dtproj_w + (size_t)i * CD * CR, dtproj_b + (size_t)i * CD, dtv);

        k_scan<<<(CB * CD) / 256, 256, 0, stream>>>(
            u, dtv, ssm, A_log + (size_t)i * CD * CN, D_param + (size_t)i * CD, yv);

        k_gate<<<nBlk, 256, 0, stream>>>(yv, proj, yg16);

        // outproj + residual: h += yg16(BT,D) x wtout(D,H)   N=256 -> NT=4
        { dim3 g(CH / 64, BT / 64);
          k_gemm_wmma<4, true><<<g, 128, 0, stream>>>(
              yg16, wtout + (size_t)i * CD * CH, h, h,
              BT, CH, CD); }
    }

    k_final<<<BT, 256, 0, stream>>>(h, normf_w, post_g, post_b, (float*)d_out);
}
